// GAT_66580583022828
// MI455X (gfx1250) — compile-verified, matrix operand order checked by reading, not run on hardware
//
#include <hip/hip_runtime.h>
#include <math.h>

// ---------------------------------------------------------------------------
// 2-layer GAT forward for MI455X (gfx1250, wave32, WMMA).
//
// GEMM1 (x[N,512] @ W1[512,128]) is bandwidth-bound (~205MB mandatory reads,
// ~9us @ 23.3 TB/s): f16-input WMMA w/ f32 accum, A staged through LDS in
// fragment order (1x global read per block instead of 8x), B pre-packed into
// WMMA fragment layout so each lane does two b128 loads per K-step.
// The runtime dominator is the edge scatter phase (~300M f32 atomics), whose
// targets (~110MB live) are L2-resident on the 192MB L2.
// ---------------------------------------------------------------------------

typedef __attribute__((ext_vector_type(16))) _Float16 v16h;
typedef __attribute__((ext_vector_type(8)))  float    v8f;

// Monotonic float->uint mapping so atomicMax(u32) orders like float.
__device__ __forceinline__ unsigned fmap(float f) {
  unsigned u = __float_as_uint(f);
  return u ^ ((u >> 31) ? 0xFFFFFFFFu : 0x80000000u);
}
__device__ __forceinline__ float funmap(unsigned u) {
  unsigned b = (u >> 31) ? (u ^ 0x80000000u) : ~u;
  return __uint_as_float(b);
}

__device__ __forceinline__ float leaky(float v) { return v > 0.f ? v : 0.2f * v; }

__device__ __forceinline__ unsigned pack2h(float lo, float hi) {
  _Float16 h0 = (_Float16)lo, h1 = (_Float16)hi;
  return ((unsigned)__builtin_bit_cast(unsigned short, h1) << 16) |
         (unsigned)__builtin_bit_cast(unsigned short, h0);
}

// WMMA 16-bit fragment k-mapping per lane (hi = lane>>4), element i in 0..15:
//   i < 8  : k = 8*hi + i
//   i >= 8 : k = 16 + 8*hi + (i-8)
__device__ __forceinline__ int frag_k(int hi, int i) {
  return (i < 8) ? (8 * hi + i) : (16 + 8 * hi + (i - 8));
}

// ---------------------------------------------------------------------------
// Pack W1[512,128] (f32, row-major KxN) into f16 WMMA B fragments:
// half index = ((kt*8 + w)*32 + lane)*16 + i  for kt=0..15, w=0..7.
// ---------------------------------------------------------------------------
__global__ __launch_bounds__(256)
void pack_w1_kernel(const float* __restrict__ W1, _Float16* __restrict__ w1pk) {
  const int kt   = blockIdx.x;          // 0..15
  const int w    = threadIdx.x >> 5;    // 0..7
  const int lane = threadIdx.x & 31;
  const int mlo  = lane & 15, hi = lane >> 4;
  const int col  = w * 16 + mlo;
  v16h b;
#pragma unroll
  for (int i = 0; i < 16; ++i)
    b[i] = (_Float16)W1[(size_t)(kt * 32 + frag_k(hi, i)) * 128 + col];
  *(v16h*)(w1pk + ((size_t)(kt * 8 + w) * 32 + lane) * 16) = b;
}

// Pack W2[128,40] into f16 fragments: kt=0..3, w=0..2 (cols clamped to 39).
__global__ __launch_bounds__(96)
void pack_w2_kernel(const float* __restrict__ W2, _Float16* __restrict__ w2pk) {
  const int kt   = blockIdx.x;          // 0..3
  const int w    = threadIdx.x >> 5;    // 0..2
  const int lane = threadIdx.x & 31;
  const int mlo  = lane & 15, hi = lane >> 4;
  const int col  = w * 16 + mlo;
  const int bcol = col < 40 ? col : 39;
  v16h b;
#pragma unroll
  for (int i = 0; i < 16; ++i)
    b[i] = (_Float16)W2[(size_t)(kt * 32 + frag_k(hi, i)) * 40 + bcol];
  *(v16h*)(w2pk + ((size_t)(kt * 3 + w) * 32 + lane) * 16) = b;
}

// ---------------------------------------------------------------------------
// GEMM1 + per-head attention dots.
// h1 = x[N x 512] @ W1[512 x 128]; wave w computes cols [16w,16w+16) == head w.
// A tile staged in LDS in fragment order; a_src/a_dst reduced out of the
// accumulator with 16-lane shfl_xor.
// C/D f32 layout: vgpr r at lane l holds (M = r + 8*(l>>4), N = l&15).
// ---------------------------------------------------------------------------
__global__ __launch_bounds__(256)
void gemm1_att_kernel(const float* __restrict__ x, const _Float16* __restrict__ w1pk,
                      const float* __restrict__ att_src1, const float* __restrict__ att_dst1,
                      float* __restrict__ h1, float* __restrict__ a_src1,
                      float* __restrict__ a_dst1, int N) {
  __shared__ unsigned ldsA[4096];   // 16 rows x 512 k as f16, fragment order

  const int row0 = blockIdx.x * 16;
  const int lane = threadIdx.x & 31;
  const int wave = threadIdx.x >> 5;   // 0..7 : column tile == head
  const int mlo  = lane & 15;
  const int hi   = lane >> 4;

  // ---- cooperative fill: u32 j = ((kt*32 + lane)*8 + p), pair (i=2p, i=2p+1)
  for (int j = threadIdx.x; j < 4096; j += 256) {
    const int fkt   = j >> 8;
    const int rem   = j & 255;
    const int flane = rem >> 3;
    const int p     = rem & 7;
    const int fml   = flane & 15, fhi = flane >> 4;
    int frow = row0 + fml; if (frow >= N) frow = N - 1;
    const int k = fkt * 32 + frag_k(fhi, 2 * p);   // k of even element; k+1 next
    const float2 f = *(const float2*)(x + (size_t)frow * 512 + k);
    ldsA[j] = pack2h(f.x, f.y);
  }
  __syncthreads();

  v8f acc = {};
  for (int kt = 0; kt < 16; ++kt) {
    v16h a = *(const v16h*)(&ldsA[((size_t)kt * 32 + lane) * 8]);
    v16h b = *(const v16h*)(w1pk + ((size_t)(kt * 8 + wave) * 32 + lane) * 16);
    acc = __builtin_amdgcn_wmma_f32_16x16x32_f16(false, a, false, b,
                                                 (short)0, acc, false, false);
  }

  const int col = wave * 16 + mlo;
  const bool full = (row0 + 16 <= N);
  if (full) {
#pragma unroll
    for (int r = 0; r < 8; ++r)
      h1[(size_t)(row0 + r + 8 * hi) * 128 + col] = acc[r];
  } else {
#pragma unroll
    for (int r = 0; r < 8; ++r) {
      int row = row0 + r + 8 * hi;
      if (row < N) h1[(size_t)row * 128 + col] = acc[r];
    }
  }

  // attention projections: head = wave, channel = mlo
  const float as = att_src1[wave * 16 + mlo];
  const float ad = att_dst1[wave * 16 + mlo];
#pragma unroll
  for (int r = 0; r < 8; ++r) {
    float vs = acc[r] * as;
    float vd = acc[r] * ad;
#pragma unroll
    for (int off = 1; off < 16; off <<= 1) {   // reduce over the 16-lane half
      vs += __shfl_xor(vs, off, 32);
      vd += __shfl_xor(vd, off, 32);
    }
    int row = row0 + r + 8 * hi;
    if (mlo == 0 && row < N) {
      a_src1[(size_t)row * 8 + wave] = vs;
      a_dst1[(size_t)row * 8 + wave] = vd;
    }
  }
}

// ---------------------------------------------------------------------------
// GEMM2: h2 = hmid[N x 128] @ W2[128 x 40]  (3 col tiles, masked to 40)
// ---------------------------------------------------------------------------
__global__ __launch_bounds__(96)
void gemm2_kernel(const float* __restrict__ hmid, const _Float16* __restrict__ w2pk,
                  float* __restrict__ h2, int N) {
  const int row0 = blockIdx.x * 16;
  const int lane = threadIdx.x & 31;
  const int wave = threadIdx.x >> 5;   // 0..2
  const int mlo  = lane & 15;
  const int hi   = lane >> 4;
  int arow = row0 + mlo; if (arow >= N) arow = N - 1;
  const int col  = wave * 16 + mlo;

  v8f acc = {};
#pragma unroll
  for (int kt = 0; kt < 4; ++kt) {
    const float* ap = hmid + (size_t)arow * 128 + kt * 32 + 8 * hi;
    const float4 t0 = *(const float4*)(ap + 0);
    const float4 t1 = *(const float4*)(ap + 4);
    const float4 t2 = *(const float4*)(ap + 16);
    const float4 t3 = *(const float4*)(ap + 20);
    v16h a;
    a[0] = (_Float16)t0.x; a[1] = (_Float16)t0.y; a[2]  = (_Float16)t0.z; a[3]  = (_Float16)t0.w;
    a[4] = (_Float16)t1.x; a[5] = (_Float16)t1.y; a[6]  = (_Float16)t1.z; a[7]  = (_Float16)t1.w;
    a[8] = (_Float16)t2.x; a[9] = (_Float16)t2.y; a[10] = (_Float16)t2.z; a[11] = (_Float16)t2.w;
    a[12]= (_Float16)t3.x; a[13]= (_Float16)t3.y; a[14] = (_Float16)t3.z; a[15] = (_Float16)t3.w;
    v16h b = *(const v16h*)(w2pk + ((size_t)(kt * 3 + wave) * 32 + lane) * 16);
    acc = __builtin_amdgcn_wmma_f32_16x16x32_f16(false, a, false, b,
                                                 (short)0, acc, false, false);
  }
  if (col < 40) {
    if (row0 + 16 <= N) {
#pragma unroll
      for (int r = 0; r < 8; ++r)
        h2[(size_t)(row0 + r + 8 * hi) * 40 + col] = acc[r];
    } else {
#pragma unroll
      for (int r = 0; r < 8; ++r) {
        int row = row0 + r + 8 * hi;
        if (row < N) h2[(size_t)row * 40 + col] = acc[r];
      }
    }
  }
}

// ---------------------------------------------------------------------------
// Layer-1 edge kernels (H=8). Edge e >= Eorig is the self-loop (e-Eorig).
// ---------------------------------------------------------------------------
__device__ __forceinline__ void edge_sd(int e, int Eorig, const int* __restrict__ src,
                                        const int* __restrict__ dst, int& s, int& d) {
  if (e < Eorig) { s = src[e]; d = dst[e]; } else { s = e - Eorig; d = s; }
}

__global__ void edge_max1_kernel(const int* __restrict__ src, const int* __restrict__ dst,
                                 const float* __restrict__ a_src, const float* __restrict__ a_dst,
                                 unsigned* __restrict__ m1, int Etot, int Eorig) {
  int e = blockIdx.x * blockDim.x + threadIdx.x;
  if (e >= Etot) return;
  int s, d; edge_sd(e, Eorig, src, dst, s, d);
  const float4* ap = (const float4*)(a_src + (size_t)s * 8);
  const float4* dp = (const float4*)(a_dst + (size_t)d * 8);
  float4 a0 = ap[0], a1 = ap[1], b0 = dp[0], b1 = dp[1];
  float ev[8] = {a0.x + b0.x, a0.y + b0.y, a0.z + b0.z, a0.w + b0.w,
                 a1.x + b1.x, a1.y + b1.y, a1.z + b1.z, a1.w + b1.w};
  unsigned* mp = m1 + (size_t)d * 8;
#pragma unroll
  for (int h = 0; h < 8; ++h) atomicMax(&mp[h], fmap(leaky(ev[h])));
}

__global__ void edge_sum1_kernel(const int* __restrict__ src, const int* __restrict__ dst,
                                 const float* __restrict__ a_src, const float* __restrict__ a_dst,
                                 const unsigned* __restrict__ m1, float* __restrict__ s1,
                                 int Etot, int Eorig) {
  int e = blockIdx.x * blockDim.x + threadIdx.x;
  if (e >= Etot) return;
  int s, d; edge_sd(e, Eorig, src, dst, s, d);
  const float4* ap = (const float4*)(a_src + (size_t)s * 8);
  const float4* dp = (const float4*)(a_dst + (size_t)d * 8);
  float4 a0 = ap[0], a1 = ap[1], b0 = dp[0], b1 = dp[1];
  float ev[8] = {a0.x + b0.x, a0.y + b0.y, a0.z + b0.z, a0.w + b0.w,
                 a1.x + b1.x, a1.y + b1.y, a1.z + b1.z, a1.w + b1.w};
  const unsigned* mp = m1 + (size_t)d * 8;
  float* sp = s1 + (size_t)d * 8;
#pragma unroll
  for (int h = 0; h < 8; ++h)
    atomicAdd(&sp[h], __expf(leaky(ev[h]) - funmap(mp[h])));
}

// one thread per (edge, head): 16-channel weighted scatter-add
__global__ void edge_agg1_kernel(const int* __restrict__ src, const int* __restrict__ dst,
                                 const float* __restrict__ a_src, const float* __restrict__ a_dst,
                                 const unsigned* __restrict__ m1, const float* __restrict__ s1,
                                 const float* __restrict__ h1, float* __restrict__ agg1,
                                 int Etot, int Eorig) {
  long long idx = (long long)blockIdx.x * blockDim.x + threadIdx.x;
  if (idx >= (long long)Etot * 8) return;
  int e = (int)(idx >> 3);
  int h = (int)(idx & 7);
  int s, d; edge_sd(e, Eorig, src, dst, s, d);
  float v = leaky(a_src[(size_t)s * 8 + h] + a_dst[(size_t)d * 8 + h]);
  float alpha = __expf(v - funmap(m1[(size_t)d * 8 + h])) / s1[(size_t)d * 8 + h];
  const float4* hp = (const float4*)(h1 + (size_t)s * 128 + h * 16);
  float* op = agg1 + (size_t)d * 128 + h * 16;
#pragma unroll
  for (int q = 0; q < 4; ++q) {
    float4 t = hp[q];
    atomicAdd(&op[q * 4 + 0], t.x * alpha);
    atomicAdd(&op[q * 4 + 1], t.y * alpha);
    atomicAdd(&op[q * 4 + 2], t.z * alpha);
    atomicAdd(&op[q * 4 + 3], t.w * alpha);
  }
}

// out1 += b1 then ELU, in place (agg1 becomes hmid)
__global__ void elu_bias_kernel(float* __restrict__ agg1, const float* __restrict__ b1,
                                long long total) {
  long long i = (long long)blockIdx.x * blockDim.x + threadIdx.x;
  if (i >= total) return;
  float v = agg1[i] + b1[(int)(i & 127)];
  agg1[i] = v > 0.f ? v : (__expf(v) - 1.f);
}

// ---------------------------------------------------------------------------
// Layer-2 (H=1, C=40)
// ---------------------------------------------------------------------------
__global__ void att2_kernel(const float* __restrict__ h2, const float* __restrict__ att_src2,
                            const float* __restrict__ att_dst2, float* __restrict__ a_src2,
                            float* __restrict__ a_dst2, int N) {
  int n = blockIdx.x * blockDim.x + threadIdx.x;
  if (n >= N) return;
  const float* hp = h2 + (size_t)n * 40;
  float s = 0.f, d = 0.f;
#pragma unroll
  for (int c = 0; c < 40; ++c) { float v = hp[c]; s += v * att_src2[c]; d += v * att_dst2[c]; }
  a_src2[n] = s; a_dst2[n] = d;
}

__global__ void edge_max2_kernel(const int* __restrict__ src, const int* __restrict__ dst,
                                 const float* __restrict__ a_src2, const float* __restrict__ a_dst2,
                                 unsigned* __restrict__ m2, int Etot, int Eorig) {
  int e = blockIdx.x * blockDim.x + threadIdx.x;
  if (e >= Etot) return;
  int s, d; edge_sd(e, Eorig, src, dst, s, d);
  atomicMax(&m2[d], fmap(leaky(a_src2[s] + a_dst2[d])));
}

__global__ void edge_sum2_kernel(const int* __restrict__ src, const int* __restrict__ dst,
                                 const float* __restrict__ a_src2, const float* __restrict__ a_dst2,
                                 const unsigned* __restrict__ m2, float* __restrict__ s2,
                                 int Etot, int Eorig) {
  int e = blockIdx.x * blockDim.x + threadIdx.x;
  if (e >= Etot) return;
  int s, d; edge_sd(e, Eorig, src, dst, s, d);
  float v = leaky(a_src2[s] + a_dst2[d]);
  atomicAdd(&s2[d], __expf(v - funmap(m2[d])));
}

__global__ void edge_agg2_kernel(const int* __restrict__ src, const int* __restrict__ dst,
                                 const float* __restrict__ a_src2, const float* __restrict__ a_dst2,
                                 const unsigned* __restrict__ m2, const float* __restrict__ s2,
                                 const float* __restrict__ h2, float* __restrict__ agg2,
                                 int Etot, int Eorig) {
  int e = blockIdx.x * blockDim.x + threadIdx.x;
  if (e >= Etot) return;
  int s, d; edge_sd(e, Eorig, src, dst, s, d);
  float v = leaky(a_src2[s] + a_dst2[d]);
  float alpha = __expf(v - funmap(m2[d])) / s2[d];
  const float4* hp = (const float4*)(h2 + (size_t)s * 40);
  float* op = agg2 + (size_t)d * 40;
#pragma unroll
  for (int q = 0; q < 10; ++q) {
    float4 t = hp[q];
    atomicAdd(&op[q * 4 + 0], t.x * alpha);
    atomicAdd(&op[q * 4 + 1], t.y * alpha);
    atomicAdd(&op[q * 4 + 2], t.z * alpha);
    atomicAdd(&op[q * 4 + 3], t.w * alpha);
  }
}

__global__ void lsm_kernel(const float* __restrict__ agg2, const float* __restrict__ b2,
                           float* __restrict__ out, int N) {
  int n = blockIdx.x * blockDim.x + threadIdx.x;
  if (n >= N) return;
  const float* ip = agg2 + (size_t)n * 40;
  float v[40];
  float mx = -3.4e38f;
#pragma unroll
  for (int c = 0; c < 40; ++c) { v[c] = ip[c] + b2[c]; mx = fmaxf(mx, v[c]); }
  float sum = 0.f;
#pragma unroll
  for (int c = 0; c < 40; ++c) sum += __expf(v[c] - mx);
  float l = mx + __logf(sum);
  float* op = out + (size_t)n * 40;
#pragma unroll
  for (int c = 0; c < 40; ++c) op[c] = v[c] - l;
}

// ---------------------------------------------------------------------------
extern "C" void kernel_launch(void* const* d_in, const int* in_sizes, int n_in,
                              void* d_out, int out_size, void* d_ws, size_t ws_size,
                              hipStream_t stream) {
  const float* x        = (const float*)d_in[0];
  const int*   ei       = (const int*)d_in[1];
  const float* W1       = (const float*)d_in[2];
  const float* att_src1 = (const float*)d_in[3];
  const float* att_dst1 = (const float*)d_in[4];
  const float* b1       = (const float*)d_in[5];
  const float* W2       = (const float*)d_in[6];
  const float* att_src2 = (const float*)d_in[7];
  const float* att_dst2 = (const float*)d_in[8];
  const float* b2       = (const float*)d_in[9];

  const int N    = in_sizes[0] / 512;
  const int E    = in_sizes[1] / 2;
  const int Etot = E + N;
  const int* srcp = ei;
  const int* dstp = ei + E;

  char* ws = (char*)d_ws;
  size_t off = 0;
  auto carve = [&](size_t bytes) -> void* {
    void* p = ws + off;
    off = (off + bytes + 255) & ~(size_t)255;
    return p;
  };
  // read-mostly buffers
  float*    h1     = (float*)carve((size_t)N * 128 * 4);
  float*    a_src1 = (float*)carve((size_t)N * 8 * 4);
  float*    a_dst1 = (float*)carve((size_t)N * 8 * 4);
  float*    h2     = (float*)carve((size_t)N * 40 * 4);
  float*    a_src2 = (float*)carve((size_t)N * 4);
  float*    a_dst2 = (float*)carve((size_t)N * 4);
  _Float16* w1pk   = (_Float16*)carve((size_t)16 * 8 * 32 * 16 * 2);  // 128KB
  _Float16* w2pk   = (_Float16*)carve((size_t)4 * 3 * 32 * 16 * 2);   // 12KB
  // contiguous zero-init region (0 is the identity for both mapped-max and sum)
  size_t zoff = off;
  unsigned* m1   = (unsigned*)carve((size_t)N * 8 * 4);
  float*    s1   = (float*)carve((size_t)N * 8 * 4);
  float*    agg1 = (float*)carve((size_t)N * 128 * 4);   // becomes hmid in-place
  unsigned* m2   = (unsigned*)carve((size_t)N * 4);
  float*    s2   = (float*)carve((size_t)N * 4);
  float*    agg2 = (float*)carve((size_t)N * 40 * 4);
  size_t zbytes = off - zoff;

  hipMemsetAsync(ws + zoff, 0, zbytes, stream);
  pack_w1_kernel<<<16, 256, 0, stream>>>(W1, w1pk);
  pack_w2_kernel<<<4, 96, 0, stream>>>(W2, w2pk);

  const int tilesM = (N + 15) / 16;
  gemm1_att_kernel<<<tilesM, 256, 0, stream>>>(x, w1pk, att_src1, att_dst1,
                                               h1, a_src1, a_dst1, N);

  const int eb = (Etot + 255) / 256;
  edge_max1_kernel<<<eb, 256, 0, stream>>>(srcp, dstp, a_src1, a_dst1, m1, Etot, E);
  edge_sum1_kernel<<<eb, 256, 0, stream>>>(srcp, dstp, a_src1, a_dst1, m1, s1, Etot, E);
  const long long eh = (long long)Etot * 8;
  edge_agg1_kernel<<<(int)((eh + 255) / 256), 256, 0, stream>>>(
      srcp, dstp, a_src1, a_dst1, m1, s1, h1, agg1, Etot, E);

  const long long tot1 = (long long)N * 128;
  elu_bias_kernel<<<(int)((tot1 + 255) / 256), 256, 0, stream>>>(agg1, b1, tot1);

  gemm2_kernel<<<tilesM, 96, 0, stream>>>(agg1, w2pk, h2, N);
  att2_kernel<<<(N + 255) / 256, 256, 0, stream>>>(h2, att_src2, att_dst2,
                                                   a_src2, a_dst2, N);

  edge_max2_kernel<<<eb, 256, 0, stream>>>(srcp, dstp, a_src2, a_dst2, m2, Etot, E);
  edge_sum2_kernel<<<eb, 256, 0, stream>>>(srcp, dstp, a_src2, a_dst2, m2, s2, Etot, E);
  edge_agg2_kernel<<<eb, 256, 0, stream>>>(srcp, dstp, a_src2, a_dst2, m2, s2,
                                           h2, agg2, Etot, E);

  lsm_kernel<<<(N + 255) / 256, 256, 0, stream>>>(agg2, b2, (float*)d_out, N);
}
